// Phi3QuantizedMLP_71700184040132
// MI455X (gfx1250) — matrix operand order repaired and checked by reading, not compiled, not run on hardware
//
#include <hip/hip_runtime.h>

typedef __attribute__((ext_vector_type(16))) _Float16 v16h;
typedef __attribute__((ext_vector_type(8)))  _Float16 v8h;
typedef __attribute__((ext_vector_type(8)))  float    v8f;

constexpr int HID = 3072;   // hidden
constexpr int ITR = 8192;   // intermediate
constexpr int TOK = 8192;   // B*S = 4*2048

constexpr int SK  = 64;     // staged K-slab (two f16 WMMA depths)
constexpr int LDK = SK + 8; // padded LDS row: 144B stride -> conflict-free b128 lane reads

static_assert(TOK % 128 == 0 && HID % 128 == 0 && ITR % 64 == 0, "tiling");

__device__ __forceinline__ v16h make_frag(const _Float16* p0, const _Float16* p1) {
  v8h lo = *(const v8h*)p0;
  v8h hi = *(const v8h*)p1;
  return __builtin_shufflevector(lo, hi, 0,1,2,3,4,5,6,7,8,9,10,11,12,13,14,15);
}

// A fragment (16x32 f16) at K-offset kk: lane (r,hsel) holds K in
// [kk+hsel*8, +8) and [kk+16+hsel*8, +8)   (ISA 7.12.2 16-bit A layout)
__device__ __forceinline__ v16h a_frag(const _Float16* sA, int row, int hsel, int kk) {
  const _Float16* b = sA + row * LDK + kk;
  return make_frag(b + hsel * 8, b + 16 + hsel * 8);
}

// B fragment (32x16 f16) at K-offset kk: lane (c,hsel) holds K in
// [kk+hsel*8, +8) and [kk+16+hsel*8, +8)   (same K order as A)
__device__ __forceinline__ v16h b_frag(const _Float16* sB, int col, int hsel, int kk) {
  const _Float16* b = sB + col * LDK + kk;
  return make_frag(b + hsel * 8, b + 16 + hsel * 8);
}

__device__ __forceinline__ v8f wmma_f16(v16h a, v16h b, v8f c) {
  return __builtin_amdgcn_wmma_f32_16x16x32_f16(false, a, false, b, (short)0, c, false, false);
}

__device__ __forceinline__ v8h cvt8i(int4 a, int4 b) {
  v8h r = { (_Float16)(float)a.x, (_Float16)(float)a.y, (_Float16)(float)a.z, (_Float16)(float)a.w,
            (_Float16)(float)b.x, (_Float16)(float)b.y, (_Float16)(float)b.z, (_Float16)(float)b.w };
  return r;
}

__device__ __forceinline__ v8h cvt8f(float4 a, float4 b) {
  v8h r = { (_Float16)a.x, (_Float16)a.y, (_Float16)a.z, (_Float16)a.w,
            (_Float16)b.x, (_Float16)b.y, (_Float16)b.z, (_Float16)b.w };
  return r;
}

// ---------------------------------------------------------------------------
// Kernel 1: fused gate_up GEMM + SwiGLU.  Block tile: 128 rows x 64 cols
// (computed for BOTH gate channel n and up channel n+ITR).  8 waves = 4x2.
// Software-pipelined: next K-slab's global loads issue before compute.
// ---------------------------------------------------------------------------
__global__ __launch_bounds__(256)
void gemm1_swiglu(const float* __restrict__ X, const int* __restrict__ Wq,
                  const float* __restrict__ Sc, _Float16* __restrict__ H) {
  __shared__ _Float16 sA [128 * LDK];
  __shared__ _Float16 sBg[ 64 * LDK];
  __shared__ _Float16 sBu[ 64 * LDK];

  const int tid  = threadIdx.x;
  const int lane = tid & 31;
  const int wid  = tid >> 5;
  const int wm   = wid >> 1;      // 0..3 : 32-row strip
  const int wn   = wid & 1;       // 0..1 : 32-col strip
  const int r    = lane & 15;
  const int hsel = lane >> 4;

  const int m0 = blockIdx.x * 128;
  const int n0 = blockIdx.y * 64;

  v8f accg[2][2] = {};
  v8f accu[2][2] = {};

  // staging coords: A = 128 rows x 64 K f32 (32 f32/thread);
  //                 B = 64 rows x 64 K int (16 int/thread, per matrix)
  const int arow = tid >> 1;            // 0..127
  const int aseg = (tid & 1) * 32;      // 0 / 32
  const int brow = tid >> 2;            // 0..63
  const int bseg = (tid & 3) * 16;      // 0/16/32/48

  const float* Xp = X  + (size_t)(m0 + arow) * HID + aseg;
  const int*   Wg = Wq + (size_t)(n0 + brow) * HID + bseg;
  const int*   Wu = Wq + (size_t)(ITR + n0 + brow) * HID + bseg;

  float4 rA[8];
  int4   rG[4], rU[4];

  auto load_slab = [&](int k0) {
    const float4* pa = (const float4*)(Xp + k0);
#pragma unroll
    for (int i = 0; i < 8; ++i) rA[i] = pa[i];
    const int4* pg = (const int4*)(Wg + k0);
#pragma unroll
    for (int i = 0; i < 4; ++i) rG[i] = pg[i];
    const int4* pu = (const int4*)(Wu + k0);
#pragma unroll
    for (int i = 0; i < 4; ++i) rU[i] = pu[i];
  };

  load_slab(0);

  for (int k0 = 0; k0 < HID; k0 += SK) {
    { // stage current slab into LDS (f32->f16; exact int8->f16)
      _Float16* dA = sA + arow * LDK + aseg;
#pragma unroll
      for (int i = 0; i < 4; ++i) *(v8h*)(dA + i * 8) = cvt8f(rA[2 * i], rA[2 * i + 1]);
      _Float16* dG = sBg + brow * LDK + bseg;
      *(v8h*)(dG)     = cvt8i(rG[0], rG[1]);
      *(v8h*)(dG + 8) = cvt8i(rG[2], rG[3]);
      _Float16* dU = sBu + brow * LDK + bseg;
      *(v8h*)(dU)     = cvt8i(rU[0], rU[1]);
      *(v8h*)(dU + 8) = cvt8i(rU[2], rU[3]);
    }
    __syncthreads();

    // issue next slab's loads now; consumed after the compute below
    if (k0 + SK < HID) {
      load_slab(k0 + SK);
      __builtin_prefetch(Wg + k0 + 2 * SK, 0, 1);   // global_prefetch_b8 (GL2)
      __builtin_prefetch(Wu + k0 + 2 * SK, 0, 1);
    }

#pragma unroll
    for (int kk = 0; kk < SK; kk += 32) {
      // load ALL fragments into distinct registers first, then batch the WMMAs
      v16h af[2], bg[2], bu[2];
      af[0] = a_frag(sA, wm * 32 + r,      hsel, kk);
      af[1] = a_frag(sA, wm * 32 + 16 + r, hsel, kk);
#pragma unroll
      for (int j = 0; j < 2; ++j) {
        const int col = wn * 32 + j * 16 + r;
        bg[j] = b_frag(sBg, col, hsel, kk);
        bu[j] = b_frag(sBu, col, hsel, kk);
      }
#pragma unroll
      for (int j = 0; j < 2; ++j) {
        accg[0][j] = wmma_f16(af[0], bg[j], accg[0][j]);
        accg[1][j] = wmma_f16(af[1], bg[j], accg[1][j]);
        accu[0][j] = wmma_f16(af[0], bu[j], accu[0][j]);
        accu[1][j] = wmma_f16(af[1], bu[j], accu[1][j]);
      }
    }
    __syncthreads();
  }

  // Epilogue: per-channel dequant scales + SwiGLU, store h as f16.
#pragma unroll
  for (int i = 0; i < 2; ++i) {
#pragma unroll
    for (int j = 0; j < 2; ++j) {
      const int col  = n0 + wn * 32 + j * 16 + r;        // gate channel
      const int rowb = m0 + wm * 32 + i * 16 + hsel * 8; // C/D layout: lane-half -> +8 rows
      const float sg = Sc[col];
      const float su = Sc[ITR + col];
#pragma unroll
      for (int v = 0; v < 8; ++v) {
        float g   = accg[i][j][v] * sg;
        float u   = accu[i][j][v] * su;
        float sil = g / (1.0f + __expf(-g));
        H[(size_t)(rowb + v) * ITR + col] = (_Float16)(u * sil);
      }
    }
  }
}

// ---------------------------------------------------------------------------
// Kernel 2: down-proj GEMM.  Block tile 128x128, 8 waves = 4x2, wave = 32x64.
// Software-pipelined like kernel 1.
// ---------------------------------------------------------------------------
__global__ __launch_bounds__(256)
void gemm2_down(const _Float16* __restrict__ H, const int* __restrict__ Wq,
                const float* __restrict__ Sc, float* __restrict__ Out) {
  __shared__ _Float16 sA[128 * LDK];
  __shared__ _Float16 sB[128 * LDK];

  const int tid  = threadIdx.x;
  const int lane = tid & 31;
  const int wid  = tid >> 5;
  const int wm   = wid >> 1;      // 0..3
  const int wn   = wid & 1;       // 0..1 : 64-col strip
  const int r    = lane & 15;
  const int hsel = lane >> 4;

  const int m0 = blockIdx.x * 128;
  const int n0 = blockIdx.y * 128;

  v8f acc[2][4] = {};

  // staging: A = 128 x 64 f16 (32 halves/thread); B = 128 x 64 int (32 int/thread)
  const int arow = tid >> 1;        // 0..127
  const int aseg = (tid & 1) * 32;  // 0 / 32

  const _Float16* Hp = H  + (size_t)(m0 + arow) * ITR + aseg;
  const int*      Wp = Wq + (size_t)(n0 + arow) * ITR + aseg;

  uint4 rA[4];
  int4  rB[8];

  auto load_slab = [&](int k0) {
    const uint4* pa = (const uint4*)(Hp + k0);
#pragma unroll
    for (int i = 0; i < 4; ++i) rA[i] = pa[i];
    const int4* pb = (const int4*)(Wp + k0);
#pragma unroll
    for (int i = 0; i < 8; ++i) rB[i] = pb[i];
  };

  load_slab(0);

  for (int k0 = 0; k0 < ITR; k0 += SK) {
    {
      _Float16* dA = sA + arow * LDK + aseg;
#pragma unroll
      for (int i = 0; i < 4; ++i) *(uint4*)(dA + i * 8) = rA[i];
      _Float16* dB = sB + arow * LDK + aseg;
#pragma unroll
      for (int i = 0; i < 4; ++i) *(v8h*)(dB + i * 8) = cvt8i(rB[2 * i], rB[2 * i + 1]);
    }
    __syncthreads();

    if (k0 + SK < ITR) {
      load_slab(k0 + SK);
      __builtin_prefetch(Wp + k0 + 2 * SK, 0, 1);
    }

#pragma unroll
    for (int kk = 0; kk < SK; kk += 32) {
      v16h af[2], bf[4];
      af[0] = a_frag(sA, wm * 32 + r,      hsel, kk);
      af[1] = a_frag(sA, wm * 32 + 16 + r, hsel, kk);
#pragma unroll
      for (int j = 0; j < 4; ++j) bf[j] = b_frag(sB, wn * 64 + j * 16 + r, hsel, kk);
#pragma unroll
      for (int j = 0; j < 4; ++j) {
        acc[0][j] = wmma_f16(af[0], bf[j], acc[0][j]);
        acc[1][j] = wmma_f16(af[1], bf[j], acc[1][j]);
      }
    }
    __syncthreads();
  }

#pragma unroll
  for (int i = 0; i < 2; ++i) {
#pragma unroll
    for (int j = 0; j < 4; ++j) {
      const int col  = n0 + wn * 64 + j * 16 + r;
      const int rowb = m0 + wm * 32 + i * 16 + hsel * 8;
      const float sd = Sc[col];
#pragma unroll
      for (int v = 0; v < 8; ++v) {
        Out[(size_t)(rowb + v) * HID + col] = acc[i][j][v] * sd;
      }
    }
  }
}

// ---------------------------------------------------------------------------
extern "C" void kernel_launch(void* const* d_in, const int* in_sizes, int n_in,
                              void* d_out, int out_size, void* d_ws, size_t ws_size,
                              hipStream_t stream) {
  (void)in_sizes; (void)n_in; (void)out_size; (void)ws_size;
  const float* x    = (const float*)d_in[0];   // [TOK, HID] f32
  const int*   w_gu = (const int*)  d_in[1];   // [2*ITR, HID] int8-valued int32
  const float* s_gu = (const float*)d_in[2];   // [2*ITR]
  const int*   w_d  = (const int*)  d_in[3];   // [HID, ITR]
  const float* s_d  = (const float*)d_in[4];   // [HID]
  float*       out  = (float*)d_out;           // [TOK, HID] f32
  _Float16*    h    = (_Float16*)d_ws;         // [TOK, ITR] f16 (128 MB scratch)

  dim3 g1(TOK / 128, ITR / 64);   // 64 x 128 blocks
  gemm1_swiglu<<<g1, 256, 0, stream>>>(x, w_gu, s_gu, h);

  dim3 g2(TOK / 128, HID / 128);  // 64 x 24 blocks
  gemm2_down<<<g2, 256, 0, stream>>>(h, w_d, s_d, out);
}
